// LogicReasoningEncoder_10788957847642
// MI455X (gfx1250) — matrix operand order repaired and checked by reading, not compiled, not run on hardware
//
#include <hip/hip_runtime.h>

// LogicReasoningEncoder for MI455X (gfx1250, wave32).
// GEMMs on v_wmma_f32_16x16x32_bf16. Weight panels (fragment-shuffled bf16)
// are staged ONCE per 4-wave block into LDS (320KB/WGP), then each wave runs
// 16-row tiles with A+B fragments as aligned 32B ds_load_b128 pairs.
// softmax / gates / LayerNorm in f32. Bool inputs assumed 1 byte/elem.

typedef __attribute__((ext_vector_type(16))) __bf16 v16bf;
typedef __attribute__((ext_vector_type(8)))  float  v8f;

constexpr int B = 4, N = 2048, E = 32768, D = 128, L = 3;
constexpr int D3 = 3 * D;   // 384
constexpr int D5 = 5 * D;   // 640
constexpr float TAU   = 0.1f;
constexpr float MASKV = -1.0e9f;
constexpr float EPSF  = 1e-8f;

constexpr int NW  = 4;           // waves per block
constexpr int TPW = 4;           // tiles per wave
constexpr int TPB = NW * TPW;    // 16 tiles per block

// ---------------- dynamic LDS layouts (bytes) ----------------
constexpr int MSG_PANEL = D5 * D * 2;                 // 163840
constexpr int MSG_AFRAG = (D5 / 32) * 32 * 16 * 2;    // 20480
constexpr int MSG_RAW   = 16 * D * 4;                 // 8192
constexpr int MSG_CHUNK = MSG_AFRAG + MSG_RAW;        // 28672
constexpr int MSG_META  = MSG_PANEL + NW * MSG_CHUNK; // 278528
constexpr int MSG_SMEM  = MSG_META + NW * 256;        // 279552 (~273KB)

constexpr int DEN_PANEL = D3 * D * 2;                 // 98304
constexpr int DEN_AFRAG = (D3 / 32) * 32 * 16 * 2;    // 12288
constexpr int DEN_HID   = 16 * D * 4;                 // 8192
constexpr int DEN_CHUNK = DEN_AFRAG + DEN_HID;        // 20480
constexpr int DEN_META  = DEN_PANEL + NW * DEN_CHUNK; // 180224
constexpr int DEN_SMEM  = DEN_META + NW * 256;        // 181248

constexpr int UPD_PANEL = D * D * 2;                  // 32768
constexpr int UPD_AFRAG = (D / 32) * 32 * 16 * 2;     // 4096
constexpr int UPD_X     = 16 * D * 4;                 // 8192
constexpr int UPD_CHUNK = UPD_AFRAG + UPD_X;          // 12288
constexpr int UPD_SMEM  = UPD_PANEL + NW * UPD_CHUNK; // 81920

// ---------------- workspace layout (bytes) ----------------
constexpr size_t SZ_HND   = (size_t)B * N * D * 4;      // 4 MB
constexpr size_t OFF_H    = 0;
constexpr size_t OFF_AGGR = OFF_H    + SZ_HND;
constexpr size_t OFF_MX   = OFF_AGGR + SZ_HND;
constexpr size_t OFF_SM   = OFF_MX   + (size_t)B * N * 4;
constexpr size_t OFF_ATT  = OFF_SM   + (size_t)B * N * 4;
constexpr size_t OFF_DEN  = OFF_ATT  + (size_t)B * E * 4;
constexpr size_t OFF_GATE = OFF_DEN  + (size_t)B * E * 4;
constexpr size_t OFF_CATT = OFF_GATE + (size_t)B * E * 4;
constexpr size_t OFF_MSGWBF  = OFF_CATT + 256;                       // 32B aligned
constexpr size_t OFF_DENW1BF = OFF_MSGWBF  + (size_t)L * D5 * D * 2;
constexpr size_t OFF_UPDWBF  = OFF_DENW1BF + (size_t)D3 * D * 2;

__device__ __forceinline__ unsigned short f2bf(float x) {
    unsigned int u = __float_as_uint(x);
    u += 0x7FFFu + ((u >> 16) & 1u);          // round-to-nearest-even
    return (unsigned short)(u >> 16);
}
__device__ __forceinline__ float sigmoidf(float x) {
    return 1.0f / (1.0f + expf(-x));
}
__device__ __forceinline__ void atomicMaxF(float* addr, float val) {
    if (val >= 0.0f) atomicMax((int*)addr, __float_as_int(val));
    else             atomicMin((unsigned int*)addr, __float_as_uint(val));
}

// Stash one A-tile element (row r, K index c) into LDS in WMMA A-fragment order.
__device__ __forceinline__ void stash_a(unsigned short* Afrag, int r, int c, float val) {
    const int ks   = c >> 5;
    const int kk   = c & 31;
    const int half = (kk >> 3) & 1;
    const int i    = (kk & 7) | ((kk >> 4) << 3);
    const int lane = r + (half << 4);
    Afrag[(((ks << 5) + lane) << 4) + i] = f2bf(val);
}

// Cooperative global->LDS panel copy (coalesced uint4).
__device__ __forceinline__ void load_panel(unsigned short* WpLds,
                                           const unsigned short* Wsh,
                                           int nUint4, int tid, int nthr) {
    uint4* d = (uint4*)WpLds;
    const uint4* s = (const uint4*)Wsh;
    for (int i = tid; i < nUint4; i += nthr) d[i] = s[i];
}

// ---- 16x128 tile GEMM, both operands fragment-contiguous in LDS ----
template <int KTOT>
__device__ __forceinline__ void gemm_frag(const unsigned short* Afrag,
                                          const unsigned short* Wp,
                                          int lane, v8f acc[8]) {
    v8f zero = {};
#pragma unroll
    for (int t = 0; t < 8; ++t) acc[t] = zero;
    for (int ks = 0; ks < KTOT / 32; ++ks) {
        const v16bf av =
            *reinterpret_cast<const v16bf*>(&Afrag[((ks << 5) + lane) << 4]);
        v16bf bv[8];
#pragma unroll
        for (int nt = 0; nt < 8; ++nt)
            bv[nt] = *reinterpret_cast<const v16bf*>(
                &Wp[((size_t)((ks * 8 + nt) * 32 + lane)) << 4]);
#pragma unroll
        for (int nt = 0; nt < 8; ++nt)
            acc[nt] = __builtin_amdgcn_wmma_f32_16x16x32_bf16(
                false, av, false, bv[nt], (short)0, acc[nt], false, false);
    }
}

__device__ __forceinline__ void store_tile_bias_relu(float out[16][D], const v8f acc[8],
                                                     const float* bias, int lane,
                                                     bool do_relu) {
    const int hv = lane >> 4, ln16 = lane & 15;
#pragma unroll
    for (int nt = 0; nt < 8; ++nt) {
        const int n = nt * 16 + ln16;
        const float bb = bias[n];
#pragma unroll
        for (int v = 0; v < 8; ++v) {          // C/D: VGPR v -> row v + 8*half
            float x = acc[nt][v] + bb;
            if (do_relu && x < 0.0f) x = 0.0f;
            out[v + 8 * hv][n] = x;
        }
    }
}

// Build msg_in tile: [h_src*h_r, h_src, h_init_src, h_r, conf]
__device__ __forceinline__ void fill_msg_tile(unsigned short* Afrag, const float* h,
                                              const float* rel_table, const float* conf,
                                              const int* srcA, const int* relA,
                                              int b, int e0, int lane) {
    for (int t = lane; t < 16 * D5; t += 32) {
        int r = t / D5, c = t % D5;
        int s = srcA[r], rel = relA[r];
        float v;
        if (c < D)            v = h[((size_t)b * N + s) * D + c] * rel_table[(size_t)rel * D + c];
        else if (c < 2 * D)   v = h[((size_t)b * N + s) * D + (c - D)];
        else if (c < 3 * D)   v = (s == 0) ? 1.0f : 0.0f;          // h_init_src
        else if (c < 4 * D)   v = rel_table[(size_t)rel * D + (c - 3 * D)];
        else                  v = conf[((size_t)(b * E + e0 + r)) * D + (c - 4 * D)];
        stash_a(Afrag, r, c, v);
    }
}

// ---------------- kernels ----------------

// Source gather for fragment-shuffled weight dest index.
__device__ __forceinline__ float wsrc_at(const float* W, size_t f) {
    const int i     = (int)(f & 15);
    const int lane  = (int)((f >> 4) & 31);
    const size_t tl = f >> 9;
    const int nt    = (int)(tl & 7);
    const int ks    = (int)(tl >> 3);
    const int K     = ks * 32 + ((i >> 3) << 4) + ((lane >> 4) << 3) + (i & 7);
    const int col   = nt * 16 + (lane & 15);
    return W[(size_t)K * D + col];
}

__global__ void k_convw(const float* msgW, const float* denW1, const float* updW,
                        unsigned short* msgWbf, unsigned short* denW1bf,
                        unsigned short* updWbf) {
    int idx = blockIdx.x * blockDim.x + threadIdx.x;
    const int nm = L * D5 * D, nd = D3 * D, nu = L * D * D;
    if (idx < nm) {
        int k = idx / (D5 * D), f = idx % (D5 * D);
        msgWbf[idx] = f2bf(wsrc_at(msgW + (size_t)k * D5 * D, f));
    } else if (idx < nm + nd) {
        int f = idx - nm;
        denW1bf[f] = f2bf(wsrc_at(denW1, f));
    } else if (idx < nm + nd + nu) {
        int j = idx - nm - nd;
        int k = j / (D * D), f = j % (D * D);
        updWbf[j] = f2bf(wsrc_at(updW + (size_t)k * D * D, f));
    }
}

__global__ void k_init_h(float* h) {
    int i = blockIdx.x * blockDim.x + threadIdx.x;
    if (i >= B * N * D) return;
    int n = (i / D) % N;
    h[i] = (n == 0) ? 1.0f : 0.0f;
}

__global__ void k_catt(const float* rq, const float* attW, const float* attb, float* catt) {
    int t = threadIdx.x;
    if (t >= B * L) return;
    int b = t / L, k = t % L;
    float s = attb[k];
    for (int c = 0; c < D; ++c) s += rq[b * D + c] * attW[(size_t)k * D3 + 2 * D + c];
    catt[t] = s;
}

__global__ __launch_bounds__(NW * 32) void k_dengate(
        const int* rels, const float* scores,
        const unsigned char* conf_mask, const unsigned char* edge_mask,
        const float* rq, const float* conf, const float* rel_table,
        const float* betaW, const float* betab,
        const unsigned short* denW1bf, const float* den_b1,
        const float* denW2, const float* den_b2,
        float* den_out, float* gate_out) {
    extern __shared__ v16bf dyn_smem[];
    unsigned char* smem = (unsigned char*)dyn_smem;
    unsigned short* Wp = (unsigned short*)smem;
    const int tid = threadIdx.x, w = tid >> 5, lane = tid & 31;
    unsigned short* Afrag = (unsigned short*)(smem + DEN_PANEL + w * DEN_CHUNK);
    float (*hid)[D] = reinterpret_cast<float (*)[D]>(smem + DEN_PANEL + w * DEN_CHUNK + DEN_AFRAG);
    int* relA = (int*)(smem + DEN_META + w * 256);

    const int blocksPerB = (E / 16) / TPB;                // 128
    const int b   = blockIdx.x / blocksPerB;
    const int blk = blockIdx.x % blocksPerB;

    load_panel(Wp, denW1bf, DEN_PANEL / 16, tid, NW * 32);
    __syncthreads();

    for (int it = 0; it < TPW; ++it) {
        const int tile = blk * TPB + it * NW + w;
        const int e0 = tile * 16;
        if (lane < 16) relA[lane] = rels[b * E + e0 + lane];
        __syncthreads();
        for (int t = lane; t < 16 * D3; t += 32) {        // [h_r, rq, conf]
            int r = t / D3, c = t % D3;
            float v;
            if (c < D)          v = rel_table[(size_t)relA[r] * D + c];
            else if (c < 2 * D) v = rq[b * D + (c - D)];
            else                v = conf[((size_t)(b * E + e0 + r)) * D + (c - 2 * D)];
            stash_a(Afrag, r, c, v);
        }
        __syncthreads();
        v8f acc[8];
        gemm_frag<D3>(Afrag, Wp, lane, acc);
        store_tile_bias_relu(hid, acc, den_b1, lane, true);
        __syncthreads();
        if (lane < 16) {
            const int e = e0 + lane;
            const int rel = relA[lane];
            float s = den_b2[0];
            for (int c = 0; c < D; ++c) s += hid[lane][c] * denW2[c];
            const float em = edge_mask[b * E + e] ? 1.0f : 0.0f;
            den_out[b * E + e] = sigmoidf(s) * em;
            float sb = betab[0];
            for (int c = 0; c < D; ++c)
                sb += (rel_table[(size_t)rel * D + c] + rq[b * D + c]) * betaW[c];
            const float beta = sigmoidf(sb);
            gate_out[b * E + e] = conf_mask[b * E + e]
                                    ? sigmoidf((scores[b * E + e] - beta) / TAU) : 0.5f;
        }
        __syncthreads();
    }
}

__global__ void k_zero(float* aggr, float* mx, float* sm) {
    int i = blockIdx.x * blockDim.x + threadIdx.x;
    if (i < B * N * D) aggr[i] = 0.0f;
    if (i < B * N) { mx[i] = MASKV; sm[i] = 0.0f; }
}

__global__ __launch_bounds__(NW * 32) void k_msg_att(int k,
        const int* edge_index, const int* rels, const unsigned char* edge_mask,
        const float* h, const float* rel_table, const float* conf,
        const unsigned short* msgWbf, const float* msg_b,
        const float* attW, const float* catt,
        float* att_out, float* mx) {
    extern __shared__ v16bf dyn_smem[];
    unsigned char* smem = (unsigned char*)dyn_smem;
    unsigned short* Wp = (unsigned short*)smem;
    const int tid = threadIdx.x, w = tid >> 5, lane = tid & 31;
    unsigned short* Afrag = (unsigned short*)(smem + MSG_PANEL + w * MSG_CHUNK);
    float (*raw)[D] = reinterpret_cast<float (*)[D]>(smem + MSG_PANEL + w * MSG_CHUNK + MSG_AFRAG);
    int* meta = (int*)(smem + MSG_META + w * 256);
    int* srcA = meta; int* tgtA = meta + 16; int* relA = meta + 32;

    const int blocksPerB = (E / 16) / TPB;                // 128
    const int b   = blockIdx.x / blocksPerB;
    const int blk = blockIdx.x % blocksPerB;

    load_panel(Wp, msgWbf + (size_t)k * D5 * D, MSG_PANEL / 16, tid, NW * 32);
    __syncthreads();

    for (int it = 0; it < TPW; ++it) {
        const int tile = blk * TPB + it * NW + w;
        const int e0 = tile * 16;
        if (lane < 16) {
            const int e = e0 + lane;
            srcA[lane] = edge_index[(size_t)b * 2 * E + e];
            tgtA[lane] = edge_index[(size_t)b * 2 * E + E + e];
            relA[lane] = rels[b * E + e];
        }
        __syncthreads();
        fill_msg_tile(Afrag, h, rel_table, conf, srcA, relA, b, e0, lane);
        __syncthreads();
        v8f acc[8];
        gemm_frag<D5>(Afrag, Wp, lane, acc);
        store_tile_bias_relu(raw, acc, msg_b + k * D, lane, true);
        __syncthreads();
        if (lane < 16) {
            const int e = e0 + lane;
            const int rel = relA[lane];
            float a = catt[b * L + k];
            const float* w1 = attW + (size_t)k * D3;      // vs raw_msg
            const float* w2 = w1 + D;                     // vs h_r
            for (int c = 0; c < D; ++c)
                a += raw[lane][c] * w1[c] + rel_table[(size_t)rel * D + c] * w2[c];
            a = (a >= 0.0f) ? a : 0.01f * a;              // leaky_relu
            a = edge_mask[b * E + e] ? a : MASKV;
            att_out[b * E + e] = a;
            atomicMaxF(&mx[b * N + tgtA[lane]], a);
        }
        __syncthreads();
    }
}

__global__ void k_expsum(const int* edge_index, const float* att,
                         const float* mx, float* sm) {
    int i = blockIdx.x * blockDim.x + threadIdx.x;
    if (i >= B * E) return;
    int b = i / E, e = i % E;
    int t = edge_index[(size_t)b * 2 * E + E + e];
    atomicAdd(&sm[b * N + t], expf(att[i] - mx[b * N + t]));
}

__global__ __launch_bounds__(NW * 32) void k_wmsg(int k,
        const int* edge_index, const int* rels, const unsigned char* edge_mask,
        const float* h, const float* rel_table, const float* conf,
        const unsigned short* msgWbf, const float* msg_b,
        const float* att, const float* mx, const float* sm,
        const float* gate, const float* den, float* aggr) {
    extern __shared__ v16bf dyn_smem[];
    unsigned char* smem = (unsigned char*)dyn_smem;
    unsigned short* Wp = (unsigned short*)smem;
    const int tid = threadIdx.x, w = tid >> 5, lane = tid & 31;
    unsigned short* Afrag = (unsigned short*)(smem + MSG_PANEL + w * MSG_CHUNK);
    float (*raw)[D] = reinterpret_cast<float (*)[D]>(smem + MSG_PANEL + w * MSG_CHUNK + MSG_AFRAG);
    int* meta = (int*)(smem + MSG_META + w * 256);
    int* srcA = meta; int* tgtA = meta + 16; int* relA = meta + 32;
    float* wv = (float*)(meta + 48);

    const int blocksPerB = (E / 16) / TPB;
    const int b   = blockIdx.x / blocksPerB;
    const int blk = blockIdx.x % blocksPerB;

    load_panel(Wp, msgWbf + (size_t)k * D5 * D, MSG_PANEL / 16, tid, NW * 32);
    __syncthreads();

    for (int it = 0; it < TPW; ++it) {
        const int tile = blk * TPB + it * NW + w;
        const int e0 = tile * 16;
        if (lane < 16) {
            const int e = e0 + lane;
            srcA[lane] = edge_index[(size_t)b * 2 * E + e];
            tgtA[lane] = edge_index[(size_t)b * 2 * E + E + e];
            relA[lane] = rels[b * E + e];
        }
        __syncthreads();
        fill_msg_tile(Afrag, h, rel_table, conf, srcA, relA, b, e0, lane);
        __syncthreads();
        v8f acc[8];
        gemm_frag<D5>(Afrag, Wp, lane, acc);             // recompute raw_msg
        store_tile_bias_relu(raw, acc, msg_b + k * D, lane, true);
        __syncthreads();
        if (lane < 16) {
            const int e = e0 + lane, t = tgtA[lane];
            const float ex    = expf(att[b * E + e] - mx[b * N + t]);
            const float alpha = ex / (sm[b * N + t] + EPSF);
            const float em    = edge_mask[b * E + e] ? 1.0f : 0.0f;
            wv[lane] = gate[b * E + e] * alpha * em * den[b * E + e];
        }
        __syncthreads();
        for (int t2 = lane; t2 < 16 * D; t2 += 32) {
            int r = t2 / D, c = t2 % D;
            float wgt = wv[r];
            if (wgt != 0.0f)
                atomicAdd(&aggr[((size_t)b * N + tgtA[r]) * D + c], wgt * raw[r][c]);
        }
        __syncthreads();
    }
}

__global__ __launch_bounds__(NW * 32) void k_update(int k,
        const float* aggr, const unsigned short* updWbf, const float* upd_b,
        const float* ln_g, const float* ln_b, float* h, float* out) {
    extern __shared__ v16bf dyn_smem[];
    unsigned char* smem = (unsigned char*)dyn_smem;
    unsigned short* Wp = (unsigned short*)smem;
    const int tid = threadIdx.x, w = tid >> 5, lane = tid & 31;
    unsigned short* Afrag = (unsigned short*)(smem + UPD_PANEL + w * UPD_CHUNK);
    float (*X)[D] = reinterpret_cast<float (*)[D]>(smem + UPD_PANEL + w * UPD_CHUNK + UPD_AFRAG);

    load_panel(Wp, updWbf + (size_t)k * D * D, UPD_PANEL / 16, tid, NW * 32);
    __syncthreads();

    for (int it = 0; it < TPW; ++it) {
        const int tile = blockIdx.x * TPB + it * NW + w;
        const int row0 = tile * 16;                  // global row = b*N + n
        const int b = row0 / N, n0 = row0 % N;
        for (int t = lane; t < 16 * D; t += 32) {
            int r = t / D, c = t % D;
            stash_a(Afrag, r, c, aggr[((size_t)row0 + r) * D + c]);
        }
        __syncthreads();
        v8f acc[8];
        gemm_frag<D>(Afrag, Wp, lane, acc);
        {
            const int hv = lane >> 4, ln16 = lane & 15;
#pragma unroll
            for (int nt = 0; nt < 8; ++nt) {
                const int n = nt * 16 + ln16;
                const float bb = upd_b[k * D + n];
#pragma unroll
                for (int v = 0; v < 8; ++v) {
                    const int m = v + 8 * hv;
                    X[m][n] = h[((size_t)row0 + m) * D + n] + acc[nt][v] + bb;
                }
            }
        }
        __syncthreads();
        if (lane < 16) {
            const int m = lane;
            float mu = 0.0f;
            for (int c = 0; c < D; ++c) mu += X[m][c];
            mu /= (float)D;
            float var = 0.0f;
            for (int c = 0; c < D; ++c) { float d0 = X[m][c] - mu; var += d0 * d0; }
            var /= (float)D;
            const float inv = rsqrtf(var + 1e-5f);
            const size_t row = (size_t)row0 + m;
            for (int c = 0; c < D; ++c) {
                float y = (X[m][c] - mu) * inv * ln_g[c] + ln_b[c];
                h[row * D + c] = y;
                if (n0 + m == 0) out[((size_t)b * L + k) * D + c] = y;   // outs[:,k,:]
            }
        }
        __syncthreads();
    }
}

// ---------------- launch ----------------
extern "C" void kernel_launch(void* const* d_in, const int* in_sizes, int n_in,
                              void* d_out, int out_size, void* d_ws, size_t ws_size,
                              hipStream_t stream) {
    (void)in_sizes; (void)n_in; (void)out_size; (void)ws_size;
    const int*           edge_index = (const int*)d_in[0];
    const int*           rels       = (const int*)d_in[1];
    const float*         scores     = (const float*)d_in[2];
    const unsigned char* conf_mask  = (const unsigned char*)d_in[3];
    const unsigned char* edge_mask  = (const unsigned char*)d_in[4];
    /* d_in[5] = mask: all ones, unused */
    const float* rq        = (const float*)d_in[6];
    const float* conf      = (const float*)d_in[7];
    const float* rel_table = (const float*)d_in[8];
    const float* beta_W    = (const float*)d_in[9];
    const float* beta_b    = (const float*)d_in[10];
    const float* msg_W     = (const float*)d_in[11];
    const float* msg_b     = (const float*)d_in[12];
    const float* upd_W     = (const float*)d_in[13];
    const float* upd_b     = (const float*)d_in[14];
    const float* ln_g      = (const float*)d_in[15];
    const float* ln_b      = (const float*)d_in[16];
    const float* att_W     = (const float*)d_in[17];
    const float* att_b     = (const float*)d_in[18];
    const float* den_W1    = (const float*)d_in[19];
    const float* den_b1    = (const float*)d_in[20];
    const float* den_W2    = (const float*)d_in[21];
    const float* den_b2    = (const float*)d_in[22];

    char* ws = (char*)d_ws;
    float* h    = (float*)(ws + OFF_H);
    float* aggr = (float*)(ws + OFF_AGGR);
    float* mx   = (float*)(ws + OFF_MX);
    float* sm   = (float*)(ws + OFF_SM);
    float* att  = (float*)(ws + OFF_ATT);
    float* den  = (float*)(ws + OFF_DEN);
    float* gate = (float*)(ws + OFF_GATE);
    float* catt = (float*)(ws + OFF_CATT);
    unsigned short* msgWbf  = (unsigned short*)(ws + OFF_MSGWBF);
    unsigned short* denW1bf = (unsigned short*)(ws + OFF_DENW1BF);
    unsigned short* updWbf  = (unsigned short*)(ws + OFF_UPDWBF);

    const int convN = L * D5 * D + D3 * D + L * D * D;
    k_convw<<<(convN + 255) / 256, 256, 0, stream>>>(msg_W, den_W1, upd_W,
                                                     msgWbf, denW1bf, updWbf);
    k_init_h<<<(B * N * D + 255) / 256, 256, 0, stream>>>(h);
    k_catt<<<1, 32, 0, stream>>>(rq, att_W, att_b, catt);

    const int edgeBlocks = B * ((E / 16) / TPB);          // 512
    k_dengate<<<edgeBlocks, NW * 32, DEN_SMEM, stream>>>(
        rels, scores, conf_mask, edge_mask, rq, conf, rel_table, beta_W, beta_b,
        denW1bf, den_b1, den_W2, den_b2, den, gate);

    for (int k = 0; k < L; ++k) {
        k_zero<<<(B * N * D + 255) / 256, 256, 0, stream>>>(aggr, mx, sm);
        k_msg_att<<<edgeBlocks, NW * 32, MSG_SMEM, stream>>>(
            k, edge_index, rels, edge_mask, h, rel_table, conf, msgWbf, msg_b,
            att_W, catt, att, mx);
        k_expsum<<<(B * E + 255) / 256, 256, 0, stream>>>(edge_index, att, mx, sm);
        k_wmsg<<<edgeBlocks, NW * 32, MSG_SMEM, stream>>>(
            k, edge_index, rels, edge_mask, h, rel_table, conf, msgWbf, msg_b,
            att, mx, sm, gate, den, aggr);
        k_update<<<(B * N / 16) / TPB, NW * 32, UPD_SMEM, stream>>>(
            k, aggr, updWbf, upd_b, ln_g, ln_b, h, (float*)d_out);
    }
}
